// EfficientAttention_9629316677973
// MI455X (gfx1250) — matrix operand order repaired and checked by reading, not compiled, not run on hardware
//
#include <hip/hip_runtime.h>

// ---------------------------------------------------------------------------
// EfficientAttention on gfx1250 (MI455X): bf16 WMMA pipeline.
//   DIM=768, HEADS=12, HEAD_DIM=64, B=8, N=4096 (M = 32768 rows)
// ---------------------------------------------------------------------------

typedef __bf16 bf16;
typedef __attribute__((ext_vector_type(16))) __bf16 v16bf;
typedef __attribute__((ext_vector_type(8)))  __bf16 v8bf;
typedef __attribute__((ext_vector_type(4)))  __bf16 v4bf;
typedef __attribute__((ext_vector_type(8)))  float  v8f;

#define DEV __device__ __forceinline__

// float -> bf16 round-to-nearest-even (bit twiddle).
DEV bf16 f2bf(float f) {
    unsigned u = __builtin_bit_cast(unsigned, f);
    unsigned r = u + 0x7FFFu + ((u >> 16) & 1u);
    unsigned short h = (unsigned short)(r >> 16);
    return __builtin_bit_cast(bf16, h);
}

// A fragment (16x32 bf16, MxK): lane L holds row M=L&15,
// K = kb..kb+7 (v0..v3) and kb+16..kb+23 (v4..v7), kb = (L>>4)*8.
DEV v16bf load_a_frag(const bf16* __restrict__ base, int ld) {
    int lane = threadIdx.x & 31;
    int m  = lane & 15;
    int kb = (lane >> 4) << 3;
    const bf16* p = base + (size_t)m * ld + kb;
    union { v16bf v; v8bf h[2]; } u;
    u.h[0] = *(const v8bf*)(p);
    u.h[1] = *(const v8bf*)(p + 16);
    return u.v;
}

// B fragment (32x16 bf16, KxN), with W stored row-major [N x K] (y = x @ W^T):
// lane L holds column N=L&15, K = kb..kb+7 and kb+16..kb+23, kb = (L>>4)*8.
DEV v16bf load_b_frag(const bf16* __restrict__ wrow_base, int ld) {
    int lane = threadIdx.x & 31;
    int n  = lane & 15;
    int kb = (lane >> 4) << 3;
    const bf16* p = wrow_base + (size_t)n * ld + kb;
    union { v16bf v; v8bf h[2]; } u;
    u.h[0] = *(const v8bf*)(p);
    u.h[1] = *(const v8bf*)(p + 16);
    return u.v;
}

DEV v8f wmma_bf(v16bf a, v16bf b, v8f c) {
    return __builtin_amdgcn_wmma_f32_16x16x32_bf16(
        /*neg_a=*/false, a, /*neg_b=*/false, b,
        /*c_mod=*/(short)0, c, /*reuse_a=*/false, /*reuse_b=*/false);
}

// ---------------------------------------------------------------------------
// fp32 -> bf16 convert, 4 elements per thread (b128 load, b64 store)
// ---------------------------------------------------------------------------
__global__ void __launch_bounds__(256)
f32_to_bf16_kernel(const float* __restrict__ in, void* __restrict__ outv, int n4) {
    int i = blockIdx.x * 256 + threadIdx.x;
    if (i >= n4) return;
    float4 f = ((const float4*)in)[i];
    v4bf o;
    o[0] = f2bf(f.x); o[1] = f2bf(f.y); o[2] = f2bf(f.z); o[3] = f2bf(f.w);
    ((v4bf*)outv)[i] = o;
}

// ---------------------------------------------------------------------------
// Y = X @ W^T + bias. X: bf16 [M x K] row-major, W: bf16 [N x K] row-major.
// One wave computes a 64x64 output tile (4x4 WMMA subtiles): 16 b128 loads per
// 16 WMMAs. Grid: (N/64, M/64/8); all 8 waves of a block share one N tile.
// MODE 0: fp32 row-major [M x N]                      (output projection)
// MODE 1: bf16 transposed per-head [B,H,64,4096]      (V projection)
// MODE 2: bf16 row-major + fused head softmax         (Q projection)
// MODE 3: fp32 transposed per-head [B,H,64,4096]      (K projection)
// ---------------------------------------------------------------------------
template<int MODE>
__global__ void __launch_bounds__(256)
gemm_xwT(const void* __restrict__ Xv, const void* __restrict__ Wv,
         const float* __restrict__ bias, void* __restrict__ outv,
         int M, int N, int K)
{
    const bf16* X = (const bf16*)Xv;
    const bf16* W = (const bf16*)Wv;

    int wv = threadIdx.x >> 5;
    int tm = (blockIdx.y * 8 + wv) << 6;   // 64-row tile
    int tn = blockIdx.x << 6;              // 64-col tile (= one head for C=768)

    const v8f vz = {0.f,0.f,0.f,0.f,0.f,0.f,0.f,0.f};
    v8f acc[4][4];
    #pragma unroll
    for (int i = 0; i < 4; ++i)
        #pragma unroll
        for (int j = 0; j < 4; ++j) acc[i][j] = vz;

    for (int k = 0; k < K; k += 32) {
        v16bf a[4];
        #pragma unroll
        for (int i = 0; i < 4; ++i)
            a[i] = load_a_frag(X + (size_t)(tm + 16 * i) * K + k, K);
        #pragma unroll
        for (int j = 0; j < 4; ++j) {
            v16bf bf = load_b_frag(W + (size_t)(tn + 16 * j) * K + k, K);
            #pragma unroll
            for (int i = 0; i < 4; ++i)
                acc[i][j] = wmma_bf(a[i], bf, acc[i][j]);
        }
    }

    int lane = threadIdx.x & 31;
    int nof  = lane & 15;
    int mof  = (lane >> 4) << 3;   // C/D layout: VGPR r -> M = mof + r, N = nof

    if (MODE == 0) {
        float* out = (float*)outv;
        #pragma unroll
        for (int i = 0; i < 4; ++i)
            #pragma unroll
            for (int j = 0; j < 4; ++j) {
                int n = tn + j * 16 + nof;
                float bb = bias[n];
                #pragma unroll
                for (int r = 0; r < 8; ++r) {
                    int m = tm + i * 16 + mof + r;
                    out[(size_t)m * N + n] = acc[i][j][r] + bb;
                }
            }
    } else if (MODE == 1 || MODE == 3) {
        // transposed per head: out[((b*12+h)*64 + d) * 4096 + m_local]
        bf16*  ob = (bf16*)outv;
        float* of = (float*)outv;
        #pragma unroll
        for (int i = 0; i < 4; ++i)
            #pragma unroll
            for (int j = 0; j < 4; ++j) {
                int n = tn + j * 16 + nof;
                int h = n >> 6, d = n & 63;
                float bb = bias[n];
                #pragma unroll
                for (int r = 0; r < 8; ++r) {
                    int m = tm + i * 16 + mof + r;
                    int b = m >> 12, ml = m & 4095;
                    size_t idx = ((size_t)(b * 12 + h) * 64 + d) * 4096 + ml;
                    if (MODE == 1) ob[idx] = f2bf(acc[i][j][r] + bb);
                    else           of[idx] = acc[i][j][r] + bb;
                }
            }
    } else {
        // MODE 2: the 64-wide n-tile == one full head; softmax over head_dim.
        // Row m lives in 16 lanes (one wave half) x 4 subtiles j.
        bf16* out = (bf16*)outv;
        #pragma unroll
        for (int i = 0; i < 4; ++i)
            #pragma unroll
            for (int r = 0; r < 8; ++r) {
                float v0 = acc[i][0][r] + bias[tn +  0 + nof];
                float v1 = acc[i][1][r] + bias[tn + 16 + nof];
                float v2 = acc[i][2][r] + bias[tn + 32 + nof];
                float v3 = acc[i][3][r] + bias[tn + 48 + nof];
                float mx = fmaxf(fmaxf(v0, v1), fmaxf(v2, v3));
                for (int off = 8; off > 0; off >>= 1)
                    mx = fmaxf(mx, __shfl_xor(mx, off, 32));
                float e0 = __expf(v0 - mx), e1 = __expf(v1 - mx);
                float e2 = __expf(v2 - mx), e3 = __expf(v3 - mx);
                float s = e0 + e1 + e2 + e3;
                for (int off = 8; off > 0; off >>= 1)
                    s += __shfl_xor(s, off, 32);
                float inv = 1.0f / s;
                int m = tm + i * 16 + mof + r;
                size_t rb = (size_t)m * N + tn;
                out[rb +  0 + nof] = f2bf(e0 * inv);
                out[rb + 16 + nof] = f2bf(e1 * inv);
                out[rb + 32 + nof] = f2bf(e2 * inv);
                out[rb + 48 + nof] = f2bf(e3 * inv);
            }
    }
}

// ---------------------------------------------------------------------------
// K softmax over the sequence: input fp32 transposed-per-head [6144 x 4096]
// (rows contiguous -> fully coalesced), output bf16 k_t same layout.
// One 256-thread block per row.
// ---------------------------------------------------------------------------
__global__ void __launch_bounds__(256)
softmax_k_rows(const float* __restrict__ Ktf, void* __restrict__ ktv)
{
    bf16* kt = (bf16*)ktv;
    const float* p = Ktf + (size_t)blockIdx.x * 4096;

    float vals[16];
    float mx = -3.0e38f;
    #pragma unroll
    for (int i = 0; i < 16; ++i) {
        vals[i] = p[threadIdx.x + i * 256];
        mx = fmaxf(mx, vals[i]);
    }
    __shared__ float red[256];
    red[threadIdx.x] = mx; __syncthreads();
    for (int s = 128; s > 0; s >>= 1) {
        if (threadIdx.x < s)
            red[threadIdx.x] = fmaxf(red[threadIdx.x], red[threadIdx.x + s]);
        __syncthreads();
    }
    mx = red[0]; __syncthreads();

    float sum = 0.f;
    #pragma unroll
    for (int i = 0; i < 16; ++i) { vals[i] = __expf(vals[i] - mx); sum += vals[i]; }
    red[threadIdx.x] = sum; __syncthreads();
    for (int s = 128; s > 0; s >>= 1) {
        if (threadIdx.x < s) red[threadIdx.x] += red[threadIdx.x + s];
        __syncthreads();
    }
    float inv = 1.0f / red[0];

    bf16* o = kt + (size_t)blockIdx.x * 4096;
    #pragma unroll
    for (int i = 0; i < 16; ++i)
        o[threadIdx.x + i * 256] = f2bf(vals[i] * inv);
}

// ---------------------------------------------------------------------------
// kv = k^T v per head with 8-way K split for parallelism (768 blocks).
// Each (head, chunk) block: 16 waves, one 16x16 WMMA tile each over K=512.
// Writes fp32 partials; kv_reduce sums, scales by 1/8, stores kvT bf16 [e,d].
// ---------------------------------------------------------------------------
#define KV_SPLIT 8
__global__ void __launch_bounds__(512)
kv_gemm_split(const void* __restrict__ ktv, const void* __restrict__ vtv,
              float* __restrict__ partial)
{
    const bf16* kt = (const bf16*)ktv;
    const bf16* vt = (const bf16*)vtv;

    int head  = blockIdx.x / KV_SPLIT;     // 0..95
    int chunk = blockIdx.x % KV_SPLIT;     // 0..7
    int w  = threadIdx.x >> 5;             // 0..15
    int td = (w >> 2) << 4;                // d tile (M)
    int te = (w & 3)  << 4;                // e tile (N)
    const bf16* kh = kt + (size_t)head * 64 * 4096;
    const bf16* vh = vt + (size_t)head * 64 * 4096;

    int k0 = chunk * (4096 / KV_SPLIT);
    v8f acc = {0.f,0.f,0.f,0.f,0.f,0.f,0.f,0.f};
    for (int k = k0; k < k0 + 4096 / KV_SPLIT; k += 32) {
        v16bf a = load_a_frag(kh + (size_t)td * 4096 + k, 4096);
        v16bf b = load_b_frag(vh + (size_t)te * 4096 + k, 4096);
        acc = wmma_bf(a, b, acc);
    }

    int lane = threadIdx.x & 31;
    int nof  = lane & 15;
    int mof  = (lane >> 4) << 3;
    float* o = partial + ((size_t)chunk * 96 + head) * 4096;
    #pragma unroll
    for (int r = 0; r < 8; ++r) {
        int d = td + mof + r, e = te + nof;
        o[e * 64 + d] = acc[r];            // already transposed [e,d]
    }
}

__global__ void __launch_bounds__(256)
kv_reduce(const float* __restrict__ partial, void* __restrict__ kvtv)
{
    bf16* kvt = (bf16*)kvtv;
    int i = blockIdx.x * 256 + threadIdx.x;    // 0 .. 96*4096-1
    if (i >= 96 * 4096) return;
    float s = 0.f;
    #pragma unroll
    for (int c = 0; c < KV_SPLIT; ++c)
        s += partial[(size_t)c * 96 * 4096 + i];
    kvt[i] = f2bf(s * 0.125f);                 // 1/sqrt(64)
}

// ---------------------------------------------------------------------------
// attn = q_sm @ kv per head: q bf16 [32768 x 768] row-major (head slice),
// kvT bf16 [96][64x64] row-major [e,d]. K=64, fully unrolled. Wave tile 32x64.
// ---------------------------------------------------------------------------
__global__ void __launch_bounds__(256)
attn_qkv(const void* __restrict__ qbv, const void* __restrict__ kvtv,
         void* __restrict__ attnv)
{
    const bf16* qb  = (const bf16*)qbv;
    const bf16* kvt = (const bf16*)kvtv;
    bf16*       at  = (bf16*)attnv;

    int wid = blockIdx.x * (blockDim.x >> 5) + (threadIdx.x >> 5);
    if (wid >= (32768 / 32) * 12) return;
    int h  = wid % 12;
    int tm = (wid / 12) << 5;
    int b  = tm >> 12;
    const bf16* kv = kvt + (size_t)(b * 12 + h) * 4096;

    const v8f vz = {0.f,0.f,0.f,0.f,0.f,0.f,0.f,0.f};
    v8f acc[2][4];
    #pragma unroll
    for (int i = 0; i < 2; ++i)
        #pragma unroll
        for (int j = 0; j < 4; ++j) acc[i][j] = vz;

    #pragma unroll
    for (int k = 0; k < 64; k += 32) {
        v16bf a0 = load_a_frag(qb + (size_t)tm        * 768 + h * 64 + k, 768);
        v16bf a1 = load_a_frag(qb + (size_t)(tm + 16) * 768 + h * 64 + k, 768);
        #pragma unroll
        for (int j = 0; j < 4; ++j) {
            v16bf bf = load_b_frag(kv + (size_t)(j * 16) * 64 + k, 64);
            acc[0][j] = wmma_bf(a0, bf, acc[0][j]);
            acc[1][j] = wmma_bf(a1, bf, acc[1][j]);
        }
    }

    int lane = threadIdx.x & 31;
    int nof  = lane & 15;
    int mof  = (lane >> 4) << 3;
    #pragma unroll
    for (int i = 0; i < 2; ++i)
        #pragma unroll
        for (int j = 0; j < 4; ++j)
            #pragma unroll
            for (int r = 0; r < 8; ++r) {
                int m = tm + i * 16 + mof + r;
                at[(size_t)m * 768 + h * 64 + j * 16 + nof] = f2bf(acc[i][j][r]);
            }
}

// ---------------------------------------------------------------------------
// Host-side orchestration
// ---------------------------------------------------------------------------
extern "C" void kernel_launch(void* const* d_in, const int* in_sizes, int n_in,
                              void* d_out, int out_size, void* d_ws, size_t ws_size,
                              hipStream_t stream) {
    (void)in_sizes; (void)n_in; (void)out_size; (void)ws_size;

    const float* x  = (const float*)d_in[0];
    const float* Wq = (const float*)d_in[1];
    const float* bq = (const float*)d_in[2];
    const float* Wk = (const float*)d_in[3];
    const float* bk = (const float*)d_in[4];
    const float* Wv = (const float*)d_in[5];
    const float* bv = (const float*)d_in[6];
    const float* Wp = (const float*)d_in[7];
    const float* bp = (const float*)d_in[8];
    float* out = (float*)d_out;

    const int M = 32768;   // B*N = 8*4096
    const int C = 768;

    // Workspace carve-up (~320 MB with reuse).
    char* w = (char*)d_ws;
    auto take = [&](size_t bytes) -> void* {
        void* r = (void*)w;
        w += (bytes + 255) & ~(size_t)255;
        return r;
    };
    void* x_bf   = take((size_t)M * C * 2);
    void* Wq_bf  = take((size_t)C * C * 2);
    void* Wk_bf  = take((size_t)C * C * 2);
    void* Wv_bf  = take((size_t)C * C * 2);
    void* Wp_bf  = take((size_t)C * C * 2);
    void* Ktf    = take((size_t)M * C * 4);            // fp32 K proj, transposed/head
    void* q_bf   = take((size_t)M * C * 2);            // softmaxed Q, bf16
    void* k_t    = take((size_t)M * C * 2);            // softmaxed K, transposed/head
    void* v_t    = take((size_t)M * C * 2);            // V, transposed/head
    void* kvPart = take((size_t)KV_SPLIT * 96 * 64 * 64 * 4);  // fp32 partials
    void* kvT    = take((size_t)96 * 64 * 64 * 2);
    void* attn   = Ktf;                                // reuse after softmax_k

    // 1) converts to bf16 (x4 vectorized)
    {
        int n4 = (M * C) / 4;
        f32_to_bf16_kernel<<<(n4 + 255) / 256, 256, 0, stream>>>(x, x_bf, n4);
        int w4 = (C * C) / 4;
        int gb = (w4 + 255) / 256;
        f32_to_bf16_kernel<<<gb, 256, 0, stream>>>(Wq, Wq_bf, w4);
        f32_to_bf16_kernel<<<gb, 256, 0, stream>>>(Wk, Wk_bf, w4);
        f32_to_bf16_kernel<<<gb, 256, 0, stream>>>(Wv, Wv_bf, w4);
        f32_to_bf16_kernel<<<gb, 256, 0, stream>>>(Wp, Wp_bf, w4);
    }

    // Big GEMM grid: 12 N-tiles x (512/8) M-tile groups; 8 waves / block.
    dim3 ggrid(C / 64, M / 64 / 8);

    // 2) Q projection + fused per-head softmax -> q_bf
    gemm_xwT<2><<<ggrid, 256, 0, stream>>>(x_bf, Wq_bf, bq, q_bf, M, C, C);
    // 3) K projection -> fp32 transposed per head
    gemm_xwT<3><<<ggrid, 256, 0, stream>>>(x_bf, Wk_bf, bk, Ktf, M, C, C);
    // 4) V projection -> bf16 transposed per head
    gemm_xwT<1><<<ggrid, 256, 0, stream>>>(x_bf, Wv_bf, bv, v_t, M, C, C);
    // 5) K softmax over sequence (coalesced rows) -> k_t bf16
    softmax_k_rows<<<96 * 64, 256, 0, stream>>>((const float*)Ktf, k_t);
    // 6) kv = k^T v per head, 8-way split-K + reduce (scale 1/8) -> kvT
    kv_gemm_split<<<96 * KV_SPLIT, 512, 0, stream>>>(k_t, v_t, (float*)kvPart);
    kv_reduce<<<(96 * 4096 + 255) / 256, 256, 0, stream>>>((const float*)kvPart, kvT);
    // 7) attn = q_sm @ kv -> bf16 (reuses Ktf region)
    attn_qkv<<<(M / 32) * 12 / 8, 256, 0, stream>>>(q_bf, kvT, attn);
    // 8) output projection -> d_out fp32
    gemm_xwT<0><<<ggrid, 256, 0, stream>>>(attn, Wp_bf, bp, out, M, C, C);
}
